// DSVDD_5248450036236
// MI455X (gfx1250) — matrix operand order repaired and checked by reading, not compiled, not run on hardware
//
#include <hip/hip_runtime.h>

// ---------------------------------------------------------------------------
// DSVDD distance-score kernel for MI455X (gfx1250, wave32).
//
// d2[b,n,m] = |phi[b,n]|^2 + |C[:,m]|^2 - 2 phi[b,n]·C[:,m]
// Keep the 6 smallest per point (fused into the GEMM epilogue; the 629 MB d2
// tensor is never materialized). Score = softmin-weighted nearest sqrt dist;
// loss = hinge terms on nearest-3 / next-3 squared distances.
//
// Matmul (40.3 GFLOP) is compute bound (27 MB working set is L2-resident).
// fp32 WMMA on CDNA5 is only 16x16x4; bf16 WMMA is 16x16x32. So we emulate
// fp32 with split-bf16: x = hi + lo, a.b ~= ah.bh + ah.bl + al.bh
// -> 12 v_wmma_f32_16x16x32_bf16 per 16x16 output tile instead of
//    32 v_wmma_f32_16x16x4_f32, on the fast matrix pipe, ~16-bit mantissa.
// C is pre-packed into A-fragment layout (hi/lo planes) so the hot loop does
// only 32B vector loads.
// ---------------------------------------------------------------------------

typedef __attribute__((ext_vector_type(16))) __bf16 v16bf;
typedef __attribute__((ext_vector_type(8)))  float  v8f;

#define BATCH  16
#define NPTS   3136
#define DDIM   128
#define MCENT  3136
#define MTILES (MCENT / 16)     // 196
#define NTILES ((BATCH * NPTS) / 16)
#define TOPK   6
#define NU_F    0.001f
#define ALPHA_F 0.1f

// ---------------------------------------------------------------- cnorm ----
__global__ __launch_bounds__(256) void cnorm_kernel(const float* __restrict__ C,
                                                    float* __restrict__ cnorm) {
    int m = blockIdx.x * blockDim.x + threadIdx.x;
    if (m >= MCENT) return;
    float s = 0.f;
    #pragma unroll 8
    for (int k = 0; k < DDIM; ++k) {
        float c = C[(size_t)k * MCENT + m];
        s += c * c;
    }
    cnorm[m] = s;
}

// ----------------------------------------------------- pack C fragments ----
// Split C into bf16 hi/lo planes, stored in WMMA 16-bit A-fragment layout:
// fragment f = (mtile*4 + chunk)*32 + lane, 8 dwords per fragment.
// A 16x32 layout: lane<16 -> M=lane, K in {8*kh+0..7} u {16+8*kh+0..7} per
// chunk (kh = lane>>4), VGPR j packs (k, k+1) as two bf16.
__global__ __launch_bounds__(256) void pack_c_kernel(const float* __restrict__ C,
                                                     unsigned int* __restrict__ Ahi,
                                                     unsigned int* __restrict__ Alo) {
    const int gid = blockIdx.x * blockDim.x + threadIdx.x;
    if (gid >= MTILES * 4 * 32 * 8) return;
    const int j  = gid & 7;
    const int ln = (gid >> 3) & 31;
    const int c  = (gid >> 8) & 3;
    const int mt = gid >> 10;
    const int m  = mt * 16 + (ln & 15);
    const int kh = ln >> 4;
    const int k0 = 32 * c + ((j < 4) ? (8 * kh + 2 * j)
                                     : (16 + 8 * kh + 2 * (j - 4)));
    const float x0 = C[(size_t)k0 * MCENT + m];
    const float x1 = C[(size_t)(k0 + 1) * MCENT + m];
    const __bf16 h0 = (__bf16)x0;
    const __bf16 h1 = (__bf16)x1;
    const __bf16 l0 = (__bf16)(x0 - (float)h0);
    const __bf16 l1 = (__bf16)(x1 - (float)h1);
    Ahi[gid] = (unsigned int)__builtin_bit_cast(unsigned short, h0) |
               ((unsigned int)__builtin_bit_cast(unsigned short, h1) << 16);
    Alo[gid] = (unsigned int)__builtin_bit_cast(unsigned short, l0) |
               ((unsigned int)__builtin_bit_cast(unsigned short, l1) << 16);
}

// --------------------------------------------------------------- main -----
// One wave32 per tile of 16 points; loop over 196 centroid tiles.
// D layout: lane L(<16) -> point n0+L, centroids m0+0..7 in acc[0..7];
// lane 16+L -> same point, centroids m0+8..15.
__global__ __launch_bounds__(256) void dsvdd_main(const float* __restrict__ phi,
                                                  const float* __restrict__ cnorm,
                                                  const unsigned int* __restrict__ Ahi_u,
                                                  const unsigned int* __restrict__ Alo_u,
                                                  const float* __restrict__ r,
                                                  float* __restrict__ out,
                                                  float* __restrict__ partial) {
    const int lane = threadIdx.x & 31;
    const int wave = threadIdx.x >> 5;
    const int tile = blockIdx.x * (blockDim.x >> 5) + wave;
    if (tile >= NTILES) return;

    const int n0 = tile * 16;         // flat point base (b*N + n)
    const int nl = lane & 15;
    const int kh = lane >> 4;

    // ---- build B fragments (2*phi split into bf16 hi/lo) + point norm.
    // B 32x16 layout: element e of the v16bf fragment maps to
    // k = 32*c + 8*kh + e (e<8) or 32*c + 16 + 8*kh + (e-8) (e>=8).
    const float* prow = phi + (size_t)(n0 + nl) * DDIM;
    v16bf bhi[4], blo[4];
    float ps = 0.f;
    #pragma unroll
    for (int c = 0; c < 4; ++c) {
        const int kb = 32 * c + 8 * kh;
        #pragma unroll
        for (int e = 0; e < 16; ++e) {
            const float px = prow[kb + e + ((e < 8) ? 0 : 8)];
            ps += px * px;
            const float x = 2.f * px;            // fold the "2*p.C" factor
            const __bf16 h = (__bf16)x;
            const __bf16 l = (__bf16)(x - (float)h);
            bhi[c][e] = h;
            blo[c][e] = l;
        }
    }
    const float pnorm = ps + __shfl_xor(ps, 16, 32);   // lanes hold K-halves

    const v16bf* Ahi = (const v16bf*)Ahi_u;
    const v16bf* Alo = (const v16bf*)Alo_u;

    // ---- running 6-smallest set (unordered) + lazy threshold
    float t[TOPK];
    #pragma unroll
    for (int i = 0; i < TOPK; ++i) t[i] = 3.0e38f;
    float tmax = 3.0e38f;

    for (int mt = 0; mt < MTILES; ++mt) {
        const size_t fb = (size_t)mt * 128 + lane;     // (mt*4+c)*32 + lane
        v8f acc0 = {};
        v8f acc1 = {};
        v8f acc2 = {};
        #pragma unroll
        for (int c = 0; c < 4; ++c) {
            const v16bf a_h = Ahi[fb + (size_t)c * 32];
            const v16bf a_l = Alo[fb + (size_t)c * 32];
            acc0 = __builtin_amdgcn_wmma_f32_16x16x32_bf16(
                false, a_h, false, bhi[c], (short)0, acc0, false, false);
            acc1 = __builtin_amdgcn_wmma_f32_16x16x32_bf16(
                false, a_h, false, blo[c], (short)0, acc1, false, false);
            acc2 = __builtin_amdgcn_wmma_f32_16x16x32_bf16(
                false, a_l, false, bhi[c], (short)0, acc2, false, false);
        }

        const int moff = mt * 16 + (kh << 3);
        #pragma unroll
        for (int i = 0; i < 8; ++i) {
            const float d2 = pnorm + cnorm[moff + i] -
                             (acc0[i] + acc1[i] + acc2[i]);
            if (d2 < tmax) {
                // insert d2, drop the max of 7 (set stays "6 smallest")
                float c = d2;
                #pragma unroll
                for (int j = TOPK - 1; j >= 0; --j) {
                    const float lo = fminf(t[j], c);
                    c = fmaxf(t[j], c);
                    t[j] = lo;
                }
                tmax = fmaxf(fmaxf(fmaxf(t[0], t[1]), fmaxf(t[2], t[3])),
                             fmaxf(t[4], t[5]));
            }
        }
    }

    // ---- merge the two centroid halves (lane L <-> lane L+16).
    float o[TOPK];
    #pragma unroll
    for (int i = 0; i < TOPK; ++i) o[i] = __shfl_xor(t[i], 16, 32);
    #pragma unroll
    for (int i = 0; i < TOPK; ++i) {
        float c = o[i];
        #pragma unroll
        for (int j = TOPK - 1; j >= 0; --j) {
            const float lo = fminf(t[j], c);
            c = fmaxf(t[j], c);
            t[j] = lo;
        }
    }

    // ---- sort 6 ascending (optimal 12-CE network)
    #define CSWAP(i, j) { float lo_ = fminf(t[i], t[j]); float hi_ = fmaxf(t[i], t[j]); t[i] = lo_; t[j] = hi_; }
    CSWAP(0, 5) CSWAP(1, 3) CSWAP(2, 4)
    CSWAP(1, 2) CSWAP(3, 4)
    CSWAP(0, 3) CSWAP(2, 5)
    CSWAP(0, 1) CSWAP(2, 3) CSWAP(4, 5)
    CSWAP(1, 2) CSWAP(3, 4)
    #undef CSWAP

    if (lane < 16) {
        const int point = n0 + lane;                 // = b*N + n

        // score: softmin over the 3 nearest sqrt distances, weight[0]*d[0]
        const float d0 = sqrtf(fmaxf(t[0], 0.f));
        const float d1 = sqrtf(fmaxf(t[1], 0.f));
        const float d2 = sqrtf(fmaxf(t[2], 0.f));
        const float e1 = expf(d0 - d1);              // d0 is the min
        const float e2 = expf(d0 - d2);
        const float w0 = 1.f / (1.f + e1 + e2);
        out[1 + point] = w0 * d0;

        // loss partial: (1/NU)*mean hinges; K==J==3 so denominators match
        const float rv = r[0];
        const float r2 = rv * rv;
        const float att = fmaxf(t[0] - r2, 0.f) + fmaxf(t[1] - r2, 0.f) +
                          fmaxf(t[2] - r2, 0.f);
        const float rep = fmaxf(r2 - t[3] - ALPHA_F, 0.f) +
                          fmaxf(r2 - t[4] - ALPHA_F, 0.f) +
                          fmaxf(r2 - t[5] - ALPHA_F, 0.f);
        const float inv = 1.f / (NU_F * (float)(BATCH * NPTS) * 3.f);
        partial[point] = (att + rep) * inv;
    }
}

// ----------------------------------------------------------- reduction ----
// Deterministic fixed-order tree reduction (no float atomics).
__global__ __launch_bounds__(256) void reduce_kernel(const float* __restrict__ partial,
                                                     float* __restrict__ out) {
    __shared__ float sm[256];
    float s = 0.f;
    for (int i = threadIdx.x; i < BATCH * NPTS; i += 256) s += partial[i];
    sm[threadIdx.x] = s;
    __syncthreads();
    for (int w = 128; w > 0; w >>= 1) {
        if ((int)threadIdx.x < w) sm[threadIdx.x] += sm[threadIdx.x + w];
        __syncthreads();
    }
    if (threadIdx.x == 0) out[0] = sm[0];
}

// ------------------------------------------------------------- launch ----
extern "C" void kernel_launch(void* const* d_in, const int* in_sizes, int n_in,
                              void* d_out, int out_size, void* d_ws, size_t ws_size,
                              hipStream_t stream) {
    const float* phi = (const float*)d_in[0];   // [16,3136,128]
    const float* C   = (const float*)d_in[1];   // [128,3136]
    const float* r   = (const float*)d_in[2];   // [1]
    float* out = (float*)d_out;                 // [0]=loss, [1..]=score (B*N)

    // workspace layout (floats/dwords):
    //   cnorm   : MCENT
    //   partial : BATCH*NPTS
    //   Ahi     : MTILES*4*32*8  (bf16-pair dwords, fragment-major, 32B aligned)
    //   Alo     : MTILES*4*32*8
    float* cnorm   = (float*)d_ws;
    float* partial = cnorm + MCENT;
    unsigned int* Ahi = (unsigned int*)(partial + BATCH * NPTS);
    unsigned int* Alo = Ahi + (size_t)MTILES * 4 * 32 * 8;

    cnorm_kernel<<<(MCENT + 255) / 256, 256, 0, stream>>>(C, cnorm);

    const int frags = MTILES * 4 * 32 * 8;      // 200704
    pack_c_kernel<<<(frags + 255) / 256, 256, 0, stream>>>(C, Ahi, Alo);

    dsvdd_main<<<(NTILES + 7) / 8, 256, 0, stream>>>(phi, cnorm, Ahi, Alo, r,
                                                     out, partial);

    reduce_kernel<<<1, 256, 0, stream>>>(partial, out);
}
